// CustomGPT2MultiHeadAttention_88347477279269
// MI455X (gfx1250) — compile-verified, matrix operand order checked by reading, not run on hardware
//
#include <hip/hip_runtime.h>

// ---------------------------------------------------------------------------
// Problem constants (from reference)
// ---------------------------------------------------------------------------
#define D_MODEL   1024
#define NUM_HEADS 16
#define HEAD_DIM  64
#define BATCH     4
#define SEQ       1024
#define SEQ_IMG   512
#define SEQ_KV    1536        // SEQ + SEQ_IMG
#define BT        (BATCH*SEQ) // 4096 query tokens
#define BTI       (BATCH*SEQ_IMG)
#define MWORDS    (SEQ_KV/32) // 48 packed mask words per query row
#define LDS_STRIDE 40         // bf16 elements per LDS row (32 + 8 pad, 80B)

typedef __attribute__((ext_vector_type(16))) __bf16 v16bf;
typedef __attribute__((ext_vector_type(8)))  __bf16 v8bf;
typedef __attribute__((ext_vector_type(8)))  float  v8f;

// ---------------------------------------------------------------------------
// helpers
// ---------------------------------------------------------------------------
__device__ __forceinline__ __bf16 f2bf(float f) { return (__bf16)f; }   // hw cvt, RTNE
__device__ __forceinline__ unsigned short bfbits(float f) {
  return __builtin_bit_cast(unsigned short, (__bf16)f);
}
__device__ __forceinline__ v16bf mk16(v8bf lo, v8bf hi) {
  v16bf r;
#pragma unroll
  for (int i = 0; i < 8; ++i) { r[i] = lo[i]; r[i + 8] = hi[i]; }
  return r;
}
#define LDF(base, off) mk16(*(const v8bf*)((base) + (off)), *(const v8bf*)((base) + (off) + 16))
#define WMMA_BF16(a, b, c) __builtin_amdgcn_wmma_f32_16x16x32_bf16(false, (a), false, (b), (short)0, (c), false, false)

// CDNA5 async global->LDS copy (16B per lane), tracked by ASYNCcnt.
// Flat LDS-aperture addresses carry the LDS byte offset in addr[31:0] (ISA 10.2),
// so truncating the shared pointer yields the VDST LDS address operand.
__device__ __forceinline__ void async_ld16(void* lds_ptr, const void* gptr) {
  unsigned loff = (unsigned)(uintptr_t)lds_ptr;
  asm volatile("global_load_async_to_lds_b128 %0, %1, off"
               :: "v"(loff), "v"(gptr) : "memory");
}
__device__ __forceinline__ void wait_async_le4() {
#if __has_builtin(__builtin_amdgcn_s_wait_asynccnt)
  __builtin_amdgcn_s_wait_asynccnt(4);
#else
  asm volatile("s_wait_asynccnt 0x4" ::: "memory");
#endif
}

// ---------------------------------------------------------------------------
// fp32 -> bf16 conversion (vectorized x4)
// ---------------------------------------------------------------------------
__global__ __launch_bounds__(256) void cvt_f32_bf16(const float* __restrict__ s,
                                                    __bf16* __restrict__ d, int n) {
  int i = (blockIdx.x * 256 + threadIdx.x) * 4;
  if (i < n) {
    float4 v = *(const float4*)(s + i);
    d[i + 0] = f2bf(v.x);
    d[i + 1] = f2bf(v.y);
    d[i + 2] = f2bf(v.z);
    d[i + 3] = f2bf(v.w);
  }
}

// ---------------------------------------------------------------------------
// pack int32 0/1 mask into per-32kv bitmask words via wave32 ballot
// ---------------------------------------------------------------------------
__global__ __launch_bounds__(256) void pack_mask(const int* __restrict__ mask,
                                                 unsigned* __restrict__ pk, int nwords) {
  const int widx = blockIdx.x * 8 + (threadIdx.x >> 5);   // one wave per word
  const int lane = threadIdx.x & 31;
  if (widx < nwords) {
    int v = mask[(size_t)widx * 32 + lane];
    unsigned long long b = __ballot(v != 0);               // wave32: low 32 bits
    if (lane == 0) pk[widx] = (unsigned)b;
  }
}

// ---------------------------------------------------------------------------
// GEMM: Y = scale * (A[M,1024] @ Bw[1024,1024]^T)   (bf16 in, f32 accumulate)
// Block tile 128x128 (8 waves as 4m x 2n, wave tile 32x64, 8 WMMAs/k-step).
// A and B k=32 slabs are staged in LDS by async global->LDS copies, double
// buffered: s_wait_asynccnt 4 (in-order completion) + barriers separate the
// producer stage from the consumer stage. 3x less L2 read traffic than
// per-wave register streaming.
// mode 0: bf16 row-major store with batch row remap ([B,rows,1024] concat)
// mode 1: bf16 transposed-V store into VT[B,H,Dh,SEQ_KV]
// mode 2: f32 row-major store (final projection -> d_out)
// ---------------------------------------------------------------------------
__global__ __launch_bounds__(256) void gemm_bf16(const __bf16* __restrict__ A,
                                                 const __bf16* __restrict__ Bw,
                                                 void* __restrict__ Yv,
                                                 int M, int mode,
                                                 int rpb, int ld_rows, int row_off,
                                                 float scale) {
  __shared__ __bf16 As[2][128][LDS_STRIDE];   // 20 KB
  __shared__ __bf16 Bs[2][128][LDS_STRIDE];   // 20 KB

  const int tid  = threadIdx.x;
  const int lane = tid & 31;
  const int lm   = lane & 15;
  const int hi8  = (lane >> 4) ? 8 : 0;

  const int block_m = (blockIdx.x >> 3) * 128;   // 8 n-blocks of 128 cols
  const int block_n = (blockIdx.x & 7) * 128;
  if (block_m >= M) return;

  const int wid    = tid >> 5;          // 0..7
  const int wave_m = (wid >> 1) * 32;   // 0,32,64,96
  const int wave_n = (wid & 1) * 64;    // 0,64

  // cooperative stage fill: 512 16B chunks each for A and B, 2 per thread
  const int frow0 = tid >> 1;                 // rows 0..127 (chunk pairs)
  const int fc0   = (tid & 1) * 2;            // chunks {0,1} or {2,3}
  auto issue_stage = [&](int buf, int k) {
#pragma unroll
    for (int j = 0; j < 2; ++j) {
      const int c = fc0 + j;                  // 16B chunk 0..3 within k-slab
      async_ld16(&As[buf][frow0][c * 8],
                 A  + (size_t)(block_m + frow0) * D_MODEL + k + c * 8);
      async_ld16(&Bs[buf][frow0][c * 8],
                 Bw + (size_t)(block_n + frow0) * D_MODEL + k + c * 8);
    }
  };

  v8f acc[2][4] = {};

  issue_stage(0, 0);      // stage 0
  issue_stage(1, 32);     // stage 1 (prefetch)
  wait_async_le4();       // stage 0 landed
  __syncthreads();

  for (int k0 = 0; k0 < D_MODEL; k0 += 32) {
    const int buf = (k0 >> 5) & 1;

    // fragments from LDS (ds_load_b128 pairs; 80B row stride avoids conflicts)
    const __bf16* ap0 = &As[buf][wave_m + lm][0];
    const __bf16* ap1 = &As[buf][wave_m + 16 + lm][0];
    const __bf16* bp  = &Bs[buf][wave_n + lm][0];
    v16bf a0 = LDF(ap0, hi8);
    v16bf a1 = LDF(ap1, hi8);
    v16bf b0 = LDF(bp,                    hi8);
    v16bf b1 = LDF(bp + 16 * LDS_STRIDE,  hi8);
    v16bf b2 = LDF(bp + 32 * LDS_STRIDE,  hi8);
    v16bf b3 = LDF(bp + 48 * LDS_STRIDE,  hi8);

    acc[0][0] = WMMA_BF16(a0, b0, acc[0][0]);
    acc[0][1] = WMMA_BF16(a0, b1, acc[0][1]);
    acc[0][2] = WMMA_BF16(a0, b2, acc[0][2]);
    acc[0][3] = WMMA_BF16(a0, b3, acc[0][3]);
    acc[1][0] = WMMA_BF16(a1, b0, acc[1][0]);
    acc[1][1] = WMMA_BF16(a1, b1, acc[1][1]);
    acc[1][2] = WMMA_BF16(a1, b2, acc[1][2]);
    acc[1][3] = WMMA_BF16(a1, b3, acc[1][3]);

    __syncthreads();                       // all waves done reading `buf`
    int kn = k0 + 64;
    if (kn >= D_MODEL) kn = 0;             // harmless wrap on last two iters
    issue_stage(buf, kn);                  // overwrite consumed buffer
    wait_async_le4();                      // other buffer's stage landed
    __syncthreads();                       // visible to all waves
  }

#pragma unroll
  for (int mi = 0; mi < 2; ++mi) {
    const int m0 = block_m + wave_m + mi * 16 + hi8;  // first of lane's 8 rows
    if (mode == 1) {
      // transposed V store: VT[((b*16+h)*64+dh)*1536 + skv], 8 consecutive skv/lane
      const int b    = m0 / rpb;
      const int skv0 = row_off + (m0 % rpb);
      const int h    = (block_n + wave_n) >> 6;   // 64-col span = one head
#pragma unroll
      for (int sub = 0; sub < 4; ++sub) {
        const int dh = sub * 16 + lm;
        __bf16* dst = (__bf16*)Yv +
            (((size_t)(b * NUM_HEADS + h) * HEAD_DIM + dh) * SEQ_KV + skv0);
        uint4 pkv;
        unsigned* pu = &pkv.x;
#pragma unroll
        for (int i = 0; i < 4; ++i) {
          unsigned lo = bfbits(acc[mi][sub][2 * i]     * scale);
          unsigned hi = bfbits(acc[mi][sub][2 * i + 1] * scale);
          pu[i] = lo | (hi << 16);
        }
        *(uint4*)dst = pkv;
      }
    } else {
      const int col0 = block_n + wave_n + lm;
#pragma unroll
      for (int r = 0; r < 8; ++r) {
        const int m   = m0 + r;
        const int b   = m / rpb;
        const int row = b * ld_rows + row_off + (m % rpb);
#pragma unroll
        for (int sub = 0; sub < 4; ++sub) {
          const int col = col0 + sub * 16;
          const float val = acc[mi][sub][r] * scale;
          if (mode == 2) ((float*)Yv)[(size_t)row * D_MODEL + col] = val;
          else           ((__bf16*)Yv)[(size_t)row * D_MODEL + col] = f2bf(val);
        }
      }
    }
  }
}

// ---------------------------------------------------------------------------
// Flash attention: one wave per (b, h, 16-query tile).
// Q [B,S,D] bf16 pre-scaled by 1/sqrt(Dh), K [B,1536,D] bf16,
// VT [B,H,64,1536] bf16, mask packed to bitmask words.
// Online softmax, P restaged through 1KB LDS for the A-layout of P@V.
// ---------------------------------------------------------------------------
__global__ __launch_bounds__(32) void attn_kernel(const __bf16* __restrict__ Q,
                                                  const __bf16* __restrict__ K,
                                                  const __bf16* __restrict__ VT,
                                                  const unsigned* __restrict__ mpk,
                                                  __bf16* __restrict__ XO) {
  __shared__ __bf16 plds[16 * 32];     // P tile: [q(16)][kv(32)]

  const int lane = threadIdx.x & 31;
  const int lm   = lane & 15;
  const int hi8  = (lane >> 4) ? 8 : 0;

  const int qt = blockIdx.x & 63;
  const int h  = (blockIdx.x >> 6) & 15;
  const int b  = blockIdx.x >> 10;

  // Q fragments for this 16-query tile (two K=32 steps over Dh=64)
  const __bf16* qptr = Q + (size_t)(b * SEQ + qt * 16 + lm) * D_MODEL + h * HEAD_DIM;
  v16bf aQ[2];
#pragma unroll
  for (int ks = 0; ks < 2; ++ks) aQ[ks] = LDF(qptr, ks * 32 + hi8);

  v8f o[4] = {};
  float mrun[8], lrun[8];
#pragma unroll
  for (int r = 0; r < 8; ++r) { mrun[r] = -3.0e38f; lrun[r] = 0.f; }

  const unsigned* mrow = mpk + (size_t)(b * SEQ + qt * 16) * MWORDS;

  for (int c = 0; c < SEQ_KV; c += 32) {
    // ---- scores: S = Q @ K^T (two 16-key tiles, two k-steps each) ----
    const __bf16* kbase0 = K + (size_t)(b * SEQ_KV + c + lm) * D_MODEL + h * HEAD_DIM;
    const __bf16* kbase1 = kbase0 + (size_t)16 * D_MODEL;
    v16bf bK00 = LDF(kbase0, hi8);
    v16bf bK01 = LDF(kbase0, 32 + hi8);
    v16bf bK10 = LDF(kbase1, hi8);
    v16bf bK11 = LDF(kbase1, 32 + hi8);
    v8f s0 = {}, s1 = {};
    s0 = WMMA_BF16(aQ[0], bK00, s0);
    s0 = WMMA_BF16(aQ[1], bK01, s0);
    s1 = WMMA_BF16(aQ[0], bK10, s1);
    s1 = WMMA_BF16(aQ[1], bK11, s1);

    // ---- mask + online softmax (rows across 16-lane halves) ----
#pragma unroll
    for (int r = 0; r < 8; ++r) {
      const unsigned mw = mrow[(size_t)(hi8 + r) * MWORDS + (c >> 5)];
      float v0 = ((mw >> lm)        & 1u) ? s0[r] : -1.0e9f;
      float v1 = ((mw >> (16 + lm)) & 1u) ? s1[r] : -1.0e9f;

      float mx = fmaxf(v0, v1);
#pragma unroll
      for (int off = 1; off < 16; off <<= 1) mx = fmaxf(mx, __shfl_xor(mx, off, 16));

      const float mnew  = fmaxf(mrun[r], mx);
      const float alpha = __expf(mrun[r] - mnew);
      const float p0 = __expf(v0 - mnew);
      const float p1 = __expf(v1 - mnew);
      float rs = p0 + p1;
#pragma unroll
      for (int off = 1; off < 16; off <<= 1) rs += __shfl_xor(rs, off, 16);

      lrun[r] = lrun[r] * alpha + rs;
      mrun[r] = mnew;
      s0[r] = p0; s1[r] = p1;
      o[0][r] = o[0][r] * alpha;
      o[1][r] = o[1][r] * alpha;
      o[2][r] = o[2][r] * alpha;
      o[3][r] = o[3][r] * alpha;
    }

    // ---- restage P (C layout) -> LDS -> A layout ----
#pragma unroll
    for (int r = 0; r < 8; ++r) {
      const int q = hi8 + r;
      plds[q * 32 + lm]      = f2bf(s0[r]);
      plds[q * 32 + 16 + lm] = f2bf(s1[r]);
    }
    asm volatile("s_wait_dscnt 0" ::: "memory");
    v16bf aP = mk16(*(const v8bf*)(plds + lm * 32 + hi8),
                    *(const v8bf*)(plds + lm * 32 + hi8 + 16));
    asm volatile("" ::: "memory");

    // ---- O += P @ V (V transposed: contiguous along kv) ----
#pragma unroll
    for (int sub = 0; sub < 4; ++sub) {
      const __bf16* vbase = VT +
          ((size_t)(b * NUM_HEADS + h) * HEAD_DIM + sub * 16 + lm) * SEQ_KV + c + hi8;
      v16bf bV = LDF(vbase, 0);
      o[sub] = WMMA_BF16(aP, bV, o[sub]);
    }
  }

  // ---- normalize and store attention output (bf16, [BT, D]) ----
  __bf16* orow = XO + (size_t)(b * SEQ + qt * 16) * D_MODEL + h * HEAD_DIM;
#pragma unroll
  for (int r = 0; r < 8; ++r) {
    const float inv = 1.0f / lrun[r];
#pragma unroll
    for (int sub = 0; sub < 4; ++sub)
      orow[(size_t)(hi8 + r) * D_MODEL + sub * 16 + lm] = f2bf(o[sub][r] * inv);
  }
}

// ---------------------------------------------------------------------------
// host orchestration
// ---------------------------------------------------------------------------
extern "C" void kernel_launch(void* const* d_in, const int* in_sizes, int n_in,
                              void* d_out, int out_size, void* d_ws, size_t ws_size,
                              hipStream_t stream) {
  const float* hs   = (const float*)d_in[0];
  const float* img  = (const float*)d_in[1];
  const int*   mask = (const int*)d_in[2];
  const float* wq   = (const float*)d_in[3];
  const float* wk   = (const float*)d_in[4];
  const float* wv   = (const float*)d_in[5];
  const float* uk   = (const float*)d_in[6];
  const float* uv   = (const float*)d_in[7];
  const float* wo   = (const float*)d_in[8];

  char* p = (char*)d_ws;
  auto carve = [&](size_t bytes) -> char* {
    char* r = p;
    p += ((bytes + 255) / 256) * 256;
    return r;
  };
  __bf16* Xh   = (__bf16*)carve((size_t)BT * D_MODEL * 2);
  __bf16* Xi   = (__bf16*)carve((size_t)BTI * D_MODEL * 2);
  __bf16* Wqb  = (__bf16*)carve((size_t)D_MODEL * D_MODEL * 2);
  __bf16* Wkb  = (__bf16*)carve((size_t)D_MODEL * D_MODEL * 2);
  __bf16* Wvb  = (__bf16*)carve((size_t)D_MODEL * D_MODEL * 2);
  __bf16* Ukb  = (__bf16*)carve((size_t)D_MODEL * D_MODEL * 2);
  __bf16* Uvb  = (__bf16*)carve((size_t)D_MODEL * D_MODEL * 2);
  __bf16* Wob  = (__bf16*)carve((size_t)D_MODEL * D_MODEL * 2);
  __bf16* Qb   = (__bf16*)carve((size_t)BT * D_MODEL * 2);
  __bf16* Kb   = (__bf16*)carve((size_t)BATCH * SEQ_KV * D_MODEL * 2);
  __bf16* VTb  = (__bf16*)carve((size_t)BATCH * SEQ_KV * D_MODEL * 2);  // [B,H,Dh,SKV]
  __bf16* XOb  = (__bf16*)carve((size_t)BT * D_MODEL * 2);
  unsigned* Mpk = (unsigned*)carve((size_t)BATCH * SEQ * MWORDS * 4);

  auto cvt = [&](const float* s, __bf16* d, int n) {
    cvt_f32_bf16<<<n / 1024, 256, 0, stream>>>(s, d, n);
  };
  cvt(hs,  Xh,  BT * D_MODEL);
  cvt(img, Xi,  BTI * D_MODEL);
  cvt(wq,  Wqb, D_MODEL * D_MODEL);
  cvt(wk,  Wkb, D_MODEL * D_MODEL);
  cvt(wv,  Wvb, D_MODEL * D_MODEL);
  cvt(uk,  Ukb, D_MODEL * D_MODEL);
  cvt(uv,  Uvb, D_MODEL * D_MODEL);
  cvt(wo,  Wob, D_MODEL * D_MODEL);

  const int nwords = BATCH * SEQ * MWORDS;            // 196608
  pack_mask<<<nwords / 8, 256, 0, stream>>>(mask, Mpk, nwords);

  // projections: blocks = (M/128 m-blocks) * (1024/128 n-blocks)
  const float qscale = 0.125f;                        // 1/sqrt(HEAD_DIM)
  gemm_bf16<<<256, 256, 0, stream>>>(Xh, Wqb, Qb,  BT,  0, SEQ,     SEQ,    0,   qscale); // Q
  gemm_bf16<<<256, 256, 0, stream>>>(Xh, Wkb, Kb,  BT,  0, SEQ,     SEQ_KV, 0,   1.0f);   // K text
  gemm_bf16<<<128, 256, 0, stream>>>(Xi, Ukb, Kb,  BTI, 0, SEQ_IMG, SEQ_KV, SEQ, 1.0f);   // K image
  gemm_bf16<<<256, 256, 0, stream>>>(Xh, Wvb, VTb, BT,  1, SEQ,     0,      0,   1.0f);   // V^T text
  gemm_bf16<<<128, 256, 0, stream>>>(Xi, Uvb, VTb, BTI, 1, SEQ_IMG, 0,      SEQ, 1.0f);   // V^T image

  // attention: B*H*(S/16) = 4*16*64 = 4096 waves
  attn_kernel<<<4096, 32, 0, stream>>>(Qb, Kb, VTb, Mpk, XOb);

  // output projection -> f32 d_out
  gemm_bf16<<<256, 256, 0, stream>>>(XOb, Wob, d_out, BT, 2, SEQ, SEQ, 0, 1.0f);
}
